// GEARS_Model_Acc_2430951490137
// MI455X (gfx1250) — matrix operand order repaired and checked by reading, not compiled, not run on hardware
//
#include <hip/hip_runtime.h>

typedef __attribute__((ext_vector_type(16))) __bf16 v16bf;
typedef __attribute__((ext_vector_type(8)))  float  v8f;

static constexpr int Gc = 5000, Bc = 128, Hc = 64, Ec = 100000;
static constexpr int Nrows = Bc * Gc;          // 640000
static constexpr int NSLOT = 512;
static constexpr int NTG   = 157;              // ceil(5000/32)
static constexpr float EPS = 1e-5f;

// ---------------- workspace layout (float offsets, all multiples of 8) -------
static constexpr int O_EMBS=0,   O_EMBQ=64,  O_BASES=128, O_BASEQ=192;
static constexpr int O_PF1S=256, O_PF1Q=320, O_PF2S=384,  O_PF2Q=448;
static constexpr int O_FUSS=512, O_FUSQ=576, O_C1S=640,   O_C1Q=704;
static constexpr int O_C2S=768,  O_C2Q=832;
static constexpr int O_DEG = 896;                       // 5000
static constexpr int O_AGG = O_DEG + 5000;              // 320000
static constexpr int O_S1S = O_AGG + 320000;            // 512*128
static constexpr int O_S1Q = O_S1S + NSLOT*128;
static constexpr int O_S2S = O_S1Q + NSLOT*128;         // 512*64
static constexpr int O_S2Q = O_S2S + NSLOT*64;
static constexpr int ZERO_N = O_S2Q + NSLOT*64;         // everything above zeroed each call
// params (written by finalize kernels)
static constexpr int O_EMBM=ZERO_N,     O_EMBR=O_EMBM+64;
static constexpr int O_PF1M=O_EMBR+64,  O_PF1R=O_PF1M+64;
static constexpr int O_PF2M=O_PF1R+64,  O_PF2R=O_PF2M+64;
static constexpr int O_C1M =O_PF2R+64,  O_C1R =O_C1M+64;
static constexpr int O_ZSC =O_C1R+64,   O_ZSH =O_ZSC+64;
static constexpr int O_TSC =O_ZSH+64,   O_TSH =O_TSC+128;
static constexpr int O_OSC =O_TSH+128,  O_OSH =O_OSC+64;
static constexpr int O_CSC =O_OSH+64,   O_CSH =O_CSC+64;
// buffers
static constexpr int O_GENER = O_CSH+64;
static constexpr int O_PERTR = O_GENER+320000;
static constexpr int O_BASE  = O_PERTR+320000;
static constexpr int O_BASEA = O_BASE +320000;
static constexpr int O_DIS   = O_BASEA+320000;          // 5000
static constexpr int O_PGE   = O_DIS+5000;              // 320000
static constexpr int O_PSUM  = O_PGE+320000;            // 8192
static constexpr int O_YPF1  = O_PSUM+8192;
static constexpr int O_YPF2  = O_YPF1+8192;
static constexpr int O_FUSED = O_YPF2+8192;
static constexpr int O_FUSEDA= O_FUSED+8192;
static constexpr int O_YC1   = O_FUSEDA+8192;
static constexpr int O_YC2   = O_YC1+8192;
static constexpr int O_W1BF  = O_YC2+8192;              // 4096 floats = 8192 bf16
static constexpr int O_W2BF  = O_W1BF+4096;
static constexpr int O_WVEC  = O_W2BF+4096;             // 640000

// ---------------- small helper kernels ---------------------------------------
__global__ void k_zero(float* p, int n) {
  int i = blockIdx.x * 256 + threadIdx.x;
  if (i < n) p[i] = 0.f;
}

// row renorm (max_norm=1) for gene_emb (+ emb BN stats) and pert_emb
__global__ void k_renorm(const float* gene, const float* pertE,
                         float* geneR, float* pertR, float* embS, float* embQ) {
  int row = blockIdx.x, h = threadIdx.x;
  bool isGene = row < Gc;
  int g = isGene ? row : row - Gc;
  const float* src = isGene ? gene : pertE;
  float v = src[g * 64 + h];
  __shared__ float sb[64];
  sb[h] = v * v;
  __syncthreads();
  if (h == 0) {
    float s = 0.f;
    for (int i = 0; i < 64; ++i) s += sb[i];
    sb[0] = fminf(1.f, 1.f / fmaxf(sqrtf(s), 1e-12f));
  }
  __syncthreads();
  float out = v * sb[0];
  if (isGene) {
    geneR[g * 64 + h] = out;
    atomicAdd(&embS[h], out);
    atomicAdd(&embQ[h], out * out);
  } else {
    pertR[g * 64 + h] = out;
  }
}

// finalize mean/rstd from sum/sumsq
__global__ void k_fin(const float* S, const float* Q, float invN,
                      float* mean, float* rstd, int C) {
  int h = threadIdx.x;
  if (h >= C) return;
  float m = S[h] * invN;
  float v = Q[h] * invN - m * m;
  mean[h] = m;
  rstd[h] = rsqrtf(v + EPS);
}

// finalize to affine scale/shift (bn folded): y*sc+sh == (y-m)*r*g+b
__global__ void k_fin_affine(const float* S, const float* Q, float invN,
                             const float* gw, const float* gb,
                             float* sc, float* sh, int C) {
  int h = threadIdx.x;
  if (h >= C) return;
  float m = S[h] * invN;
  float v = Q[h] * invN - m * m;
  float r = rsqrtf(v + EPS);
  sc[h] = r * gw[h];
  sh[h] = gb[h] - m * r * gw[h];
}

__global__ void k_fin_slots(const float* SS, const float* QQ, int C, float invN,
                            const float* gw, const float* gb, float* sc, float* sh) {
  int n = threadIdx.x;
  if (n >= C) return;
  float S = 0.f, Q = 0.f;
  for (int s = 0; s < NSLOT; ++s) { S += SS[s * C + n]; Q += QQ[s * C + n]; }
  float m = S * invN;
  float v = Q * invN - m * m;
  float r = rsqrtf(v + EPS);
  sc[n] = r * gw[n];
  sh[n] = gb[n] - m * r * gw[n];
}

// base = relu(bn(geneR)); accumulate base stats for bnp decomposition
__global__ void k_base(const float* geneR, const float* mean, const float* rstd,
                       const float* gg, const float* gb,
                       float* base, float* baseS, float* baseQ) {
  int g = blockIdx.x, h = threadIdx.x;
  float v = (geneR[g * 64 + h] - mean[h]) * rstd[h] * gg[h] + gb[h];
  v = fmaxf(v, 0.f);
  base[g * 64 + h] = v;
  atomicAdd(&baseS[h], v);
  atomicAdd(&baseQ[h], v * v);
}

__global__ void k_deg(const int* dst, const float* ew, float* deg) {
  int e = blockIdx.x * 256 + threadIdx.x;
  if (e < Ec) atomicAdd(&deg[dst[e]], ew[e]);
}

__global__ void k_dis(const float* deg, float* dis) {
  int g = blockIdx.x * 256 + threadIdx.x;
  if (g < Gc) dis[g] = rsqrtf(deg[g] + 1.0f);   // +1 self-loop weight
}

__global__ void k_edge(const int* src, const int* dst, const float* ew,
                       const float* dis, const float* pertR, float* agg) {
  int e = blockIdx.x, h = threadIdx.x;
  int s = src[e], d = dst[e];
  float nrm = dis[s] * ew[e] * dis[d];
  atomicAdd(&agg[d * 64 + h], nrm * pertR[s * 64 + h]);
}

// pge = (agg + selfloop) @ sg_W.T + sg_b
__global__ void k_pge(const float* agg, const float* pertR, const float* dis,
                      const float* W, const float* bvec, float* pge) {
  int g = blockIdx.x, h = threadIdx.x;
  __shared__ float x[64];
  float dsq = dis[g] * dis[g];
  x[h] = agg[g * 64 + h] + pertR[g * 64 + h] * dsq;
  __syncthreads();
  float acc = bvec[h];
  for (int k = 0; k < 64; ++k) acc += x[k] * W[h * 64 + k];
  pge[g * 64 + h] = acc;
}

__global__ void k_psum(const int* pert, const float* pge, float* psum) {
  int b = blockIdx.x, h = threadIdx.x;
  float acc = 0.f;
  for (int g = 0; g < Gc; ++g)
    if (pert[b * Gc + g]) acc += pge[g * 64 + h];
  psum[b * 64 + h] = acc;
}

__global__ void k_pf1(const float* psum, const float* W, const float* bvec,
                      float* y, float* S, float* Q) {
  int b = blockIdx.x, h = threadIdx.x;
  __shared__ float x[64];
  x[h] = psum[b * 64 + h];
  __syncthreads();
  float acc = bvec[h];
  for (int k = 0; k < 64; ++k) acc += x[k] * W[h * 64 + k];
  y[b * 64 + h] = acc;
  atomicAdd(&S[h], acc);
  atomicAdd(&Q[h], acc * acc);
}

// y2 = relu(bn(y1)) @ W.T + b  (64->64) with stats
__global__ void k_bnrelu_gemv(const float* yin, const float* mean, const float* rstd,
                              const float* gw, const float* gb, const float* W,
                              const float* bvec, float* yout, float* S, float* Q) {
  int b = blockIdx.x, h = threadIdx.x;
  __shared__ float x[64];
  float t = (yin[b * 64 + h] - mean[h]) * rstd[h] * gw[h] + gb[h];
  x[h] = fmaxf(t, 0.f);
  __syncthreads();
  float acc = bvec[h];
  for (int k = 0; k < 64; ++k) acc += x[k] * W[h * 64 + k];
  yout[b * 64 + h] = acc;
  atomicAdd(&S[h], acc);
  atomicAdd(&Q[h], acc * acc);
}

__global__ void k_bn_apply_stats(const float* yin, const float* mean, const float* rstd,
                                 const float* gw, const float* gb,
                                 float* out, float* S, float* Q) {
  int b = blockIdx.x, h = threadIdx.x;
  float v = (yin[b * 64 + h] - mean[h]) * rstd[h] * gw[h] + gb[h];
  out[b * 64 + h] = v;
  atomicAdd(&S[h], v);
  atomicAdd(&Q[h], v * v);
}

// bnp stats from decomposition: base over G, fused over B (cross-cov == 0)
__global__ void k_zparams(const float* baseS, const float* baseQ,
                          const float* fusS, const float* fusQ,
                          const float* bg, const float* bb, float* zsc, float* zsh) {
  int h = threadIdx.x;
  float mb = baseS[h] / (float)Gc, mf = fusS[h] / (float)Bc;
  float ex2 = baseQ[h] / (float)Gc + 2.f * mb * mf + fusQ[h] / (float)Bc;
  float mean = mb + mf;
  float var = ex2 - mean * mean;
  float r = rsqrtf(var + EPS);
  zsc[h] = r * bg[h];
  zsh[h] = bb[h] - mean * r * bg[h];
}

__global__ void k_prepA(const float* base, const float* fused,
                        const float* zsc, const float* zsh,
                        float* baseA, float* fusedA) {
  int r = blockIdx.x, h = threadIdx.x;
  if (r < Gc) baseA[r * 64 + h] = base[r * 64 + h] * zsc[h];
  else { int b = r - Gc; fusedA[b * 64 + h] = fused[b * 64 + h] * zsc[h] + zsh[h]; }
}

__global__ void k_wconv(const float* W1f, const float* W2f, __bf16* W1, __bf16* W2) {
  int i = blockIdx.x * 256 + threadIdx.x;
  if (i < 8192) W1[i] = (__bf16)W1f[i];
  else if (i < 16384) W2[i - 8192] = (__bf16)W2f[i - 8192];
}

// ---------------- WMMA fragment builders (CDNA5 16-bit layouts) --------------
__device__ __forceinline__ v16bf frag_from_f32(const float* row, int kbase) {
  v16bf a;
#pragma unroll
  for (int e = 0; e < 8; ++e)  a[e] = (__bf16)row[kbase + e];
#pragma unroll
  for (int e = 8; e < 16; ++e) a[e] = (__bf16)row[kbase + 8 + e];
  return a;
}
__device__ __forceinline__ v16bf frag_from_bf(const __bf16* row, int kbase) {
  v16bf a;
#pragma unroll
  for (int e = 0; e < 8; ++e)  a[e] = row[kbase + e];
#pragma unroll
  for (int e = 8; e < 16; ++e) a[e] = row[kbase + 8 + e];
  return a;
}
// B (K x N): lanes 0-15 hold K=kt*32+{0..7,16..23}, lanes 16-31 hold K=kt*32+{8..15,24..31}; N=lane&15
__device__ __forceinline__ v16bf loadB(const __bf16* W, int ntile, int ktile,
                                       int ldk, int lane) {
  int n = ntile * 16 + (lane & 15);
  int kb = ktile * 32 + ((lane >> 4) * 8);
  v16bf r;
#pragma unroll
  for (int e = 0; e < 8; ++e)  r[e] = W[n * ldk + kb + e];
#pragma unroll
  for (int e = 8; e < 16; ++e) r[e] = W[n * ldk + kb + 8 + e];
  return r;
}

// ---------------- fused recovery MLP (3-pass, recompute-over-store) ----------
// PHASE 0: GEMM1, accumulate y1 stats
// PHASE 1: GEMM1 -> bn+relu -> GEMM2, accumulate y2 stats
// PHASE 2: full chain -> bn -> per-gene head -> wvec[B,G]
template <int PHASE>
__global__ __launch_bounds__(256)
void k_mlp(const float* baseA, const float* fusedA,
           const __bf16* W1, const __bf16* W2,
           const float* b1, const float* tsc, const float* tsh,
           const float* b2, const float* osc, const float* osh,
           const float* iw1, const float* ib1,
           float* s1S, float* s1Q, float* s2S, float* s2Q, float* wvec) {
  __shared__ float  zn[32][64];
  __shared__ __bf16 T[32][128];
  __shared__ float  O[32][64];
  __shared__ float  red[32][8];

  const int tid = threadIdx.x;
  const int tg = blockIdx.x, b = blockIdx.y;
  const int g0 = tg * 32;

  // zn = relu(baseA[g] + fusedA[b])  (affine-folded bn_pert_base)
#pragma unroll
  for (int i = 0; i < 8; ++i) {
    int idx = tid + 256 * i;
    int m = idx >> 6, h = idx & 63;
    int g = g0 + m;
    zn[m][h] = (g < Gc) ? fmaxf(baseA[g * 64 + h] + fusedA[b * 64 + h], 0.f) : 0.f;
  }
  __syncthreads();

  const int wv = tid >> 5, lane = tid & 31;
  const int mt = wv >> 2, q = wv & 3;           // mt: 0/1 (M-tile), q: 0..3
  const int hi = lane >> 4, nl = lane & 15;
  const int slot = (b * NTG + tg) & (NSLOT - 1);

  const float* arow = &zn[mt * 16 + nl][0];
  v16bf a0 = frag_from_f32(arow, 0 + hi * 8);
  v16bf a1 = frag_from_f32(arow, 32 + hi * 8);

#pragma unroll
  for (int p = 0; p < 2; ++p) {
    int nt = q * 2 + p;                         // N-tile of 128-wide GEMM1
    v8f c = {};
    c = __builtin_amdgcn_wmma_f32_16x16x32_bf16(false, a0, false,
            loadB(W1, nt, 0, 64, lane), (short)0, c, false, false);
    c = __builtin_amdgcn_wmma_f32_16x16x32_bf16(false, a1, false,
            loadB(W1, nt, 1, 64, lane), (short)0, c, false, false);
    int n = nt * 16 + nl;
    float bias = b1[n];
    if (PHASE == 0) {
      float s = 0.f, qq = 0.f;
#pragma unroll
      for (int j = 0; j < 8; ++j) {
        int g = g0 + mt * 16 + j + 8 * hi;
        float v = c[j] + bias;
        bool ok = g < Gc;
        s += ok ? v : 0.f;
        qq += ok ? v * v : 0.f;
      }
      atomicAdd(&s1S[slot * 128 + n], s);
      atomicAdd(&s1Q[slot * 128 + n], qq);
    } else {
      float sc = tsc[n], sh = tsh[n];
#pragma unroll
      for (int j = 0; j < 8; ++j) {
        float t = fmaxf((c[j] + bias) * sc + sh, 0.f);
        T[mt * 16 + j + 8 * hi][n] = (__bf16)t;
      }
    }
  }
  if (PHASE == 0) return;
  __syncthreads();

  const __bf16* a2row = &T[mt * 16 + nl][0];
  v8f c2 = {};
#pragma unroll
  for (int kt = 0; kt < 4; ++kt) {
    v16bf a2 = frag_from_bf(a2row, kt * 32 + hi * 8);
    c2 = __builtin_amdgcn_wmma_f32_16x16x32_bf16(false, a2, false,
             loadB(W2, q, kt, 128, lane), (short)0, c2, false, false);
  }
  int n2 = q * 16 + nl;
  float bias2 = b2[n2];

  if (PHASE == 1) {
    float s = 0.f, qq = 0.f;
#pragma unroll
    for (int j = 0; j < 8; ++j) {
      int g = g0 + mt * 16 + j + 8 * hi;
      float v = c2[j] + bias2;
      bool ok = g < Gc;
      s += ok ? v : 0.f;
      qq += ok ? v * v : 0.f;
    }
    atomicAdd(&s2S[slot * 64 + n2], s);
    atomicAdd(&s2Q[slot * 64 + n2], qq);
    return;
  }

  // PHASE 2: apply final bn, per-gene linear head
  {
    float sc = osc[n2], sh = osh[n2];
#pragma unroll
    for (int j = 0; j < 8; ++j)
      O[mt * 16 + j + 8 * hi][n2] = (c2[j] + bias2) * sc + sh;
  }
  __syncthreads();
  int m = tid >> 3, part = tid & 7;
  int g = g0 + m;
  float pp = 0.f;
  if (g < Gc) {
#pragma unroll
    for (int h = part; h < 64; h += 8) pp += O[m][h] * iw1[g * 64 + h];
  }
  red[m][part] = pp;
  __syncthreads();
  if (part == 0 && g < Gc) {
    float s = red[m][0] + red[m][1] + red[m][2] + red[m][3] +
              red[m][4] + red[m][5] + red[m][6] + red[m][7];
    wvec[b * Gc + g] = s + ib1[g];
  }
}

// ---------------- cross_gene_state + final -----------------------------------
__global__ void k_cross1(const float* wvec, const float* W /*[64][5000]*/,
                         const float* bvec, float* y, float* S, float* Q) {
  int b = blockIdx.x, t = threadIdx.x;
  int h = t & 63, sl = t >> 6;
  float acc = 0.f;
  for (int g = sl * 1250; g < (sl + 1) * 1250; ++g)
    acc += wvec[b * Gc + g] * W[h * Gc + g];
  __shared__ float red[4][64];
  red[sl][h] = acc;
  __syncthreads();
  if (sl == 0) {
    float v = red[0][h] + red[1][h] + red[2][h] + red[3][h] + bvec[h];
    y[b * 64 + h] = v;
    atomicAdd(&S[h], v);
    atomicAdd(&Q[h], v * v);
  }
}

__global__ void k_final(const float* wvec, const float* yc2,
                        const float* csc, const float* csh,
                        const float* iw2, const float* ib2,
                        const float* expr, float* out) {
  int b = blockIdx.x, t = threadIdx.x;
  __shared__ float cr[64];
  if (t < 64) cr[t] = yc2[b * 64 + t] * csc[t] + csh[t];
  __syncthreads();
  for (int g = t; g < Gc; g += 256) {
    const float* w2 = iw2 + g * 65;
    float acc = wvec[b * Gc + g] * w2[0] + ib2[g] + expr[b * Gc + g];
    for (int h = 0; h < 64; ++h) acc += cr[h] * w2[1 + h];
    out[b * Gc + g] = acc;
  }
}

// ---------------- launcher ----------------------------------------------------
extern "C" void kernel_launch(void* const* d_in, const int* in_sizes, int n_in,
                              void* d_out, int out_size, void* d_ws, size_t ws_size,
                              hipStream_t stream) {
  const float* expr     = (const float*)d_in[0];
  const int*   pert     = (const int*)  d_in[1];
  const int*   eidx     = (const int*)  d_in[2];
  const float* ew       = (const float*)d_in[3];
  const float* gene_emb = (const float*)d_in[4];
  const float* pert_emb = (const float*)d_in[5];
  const float* bn_emb_g = (const float*)d_in[6];
  const float* bn_emb_b = (const float*)d_in[7];
  const float* sg_W     = (const float*)d_in[8];
  const float* sg_b     = (const float*)d_in[9];
  const float* pf_W1 = (const float*)d_in[10]; const float* pf_b1 = (const float*)d_in[11];
  const float* pf_g1 = (const float*)d_in[12]; const float* pf_be1= (const float*)d_in[13];
  const float* pf_W2 = (const float*)d_in[14]; const float* pf_b2 = (const float*)d_in[15];
  const float* pf_g2 = (const float*)d_in[16]; const float* pf_be2= (const float*)d_in[17];
  const float* bnp_g = (const float*)d_in[18]; const float* bnp_b = (const float*)d_in[19];
  const float* r_W1  = (const float*)d_in[20]; const float* r_b1  = (const float*)d_in[21];
  const float* r_g1  = (const float*)d_in[22]; const float* r_be1 = (const float*)d_in[23];
  const float* r_W2  = (const float*)d_in[24]; const float* r_b2  = (const float*)d_in[25];
  const float* r_g2  = (const float*)d_in[26]; const float* r_be2 = (const float*)d_in[27];
  const float* indv_w1 = (const float*)d_in[28]; const float* indv_b1 = (const float*)d_in[29];
  const float* c_W1  = (const float*)d_in[30]; const float* c_b1  = (const float*)d_in[31];
  const float* c_g1  = (const float*)d_in[32]; const float* c_be1 = (const float*)d_in[33];
  const float* c_W2  = (const float*)d_in[34]; const float* c_b2  = (const float*)d_in[35];
  const float* c_g2  = (const float*)d_in[36]; const float* c_be2 = (const float*)d_in[37];
  const float* indv_w2 = (const float*)d_in[38]; const float* indv_b2 = (const float*)d_in[39];

  const int* src = eidx;
  const int* dst = eidx + Ec;

  float* ws = (float*)d_ws;
  __bf16* W1bf = (__bf16*)(ws + O_W1BF);
  __bf16* W2bf = (__bf16*)(ws + O_W2BF);
  float* out = (float*)d_out;

  // 0) zero accumulators
  k_zero<<<(ZERO_N + 255) / 256, 256, 0, stream>>>(ws, ZERO_N);

  // 1) embedding renorm + emb BN stats; base = relu(bn(geneR))
  k_renorm<<<2 * Gc, 64, 0, stream>>>(gene_emb, pert_emb,
      ws + O_GENER, ws + O_PERTR, ws + O_EMBS, ws + O_EMBQ);
  k_fin<<<1, 64, 0, stream>>>(ws + O_EMBS, ws + O_EMBQ, 1.f / Gc,
      ws + O_EMBM, ws + O_EMBR, 64);
  k_base<<<Gc, 64, 0, stream>>>(ws + O_GENER, ws + O_EMBM, ws + O_EMBR,
      bn_emb_g, bn_emb_b, ws + O_BASE, ws + O_BASES, ws + O_BASEQ);

  // 2) SGConv on GO graph
  k_deg<<<(Ec + 255) / 256, 256, 0, stream>>>(dst, ew, ws + O_DEG);
  k_dis<<<(Gc + 255) / 256, 256, 0, stream>>>(ws + O_DEG, ws + O_DIS);
  k_edge<<<Ec, 64, 0, stream>>>(src, dst, ew, ws + O_DIS, ws + O_PERTR, ws + O_AGG);
  k_pge<<<Gc, 64, 0, stream>>>(ws + O_AGG, ws + O_PERTR, ws + O_DIS, sg_W, sg_b, ws + O_PGE);

  // 3) psum + pert_fuse MLP -> fused[B,64]
  k_psum<<<Bc, 64, 0, stream>>>(pert, ws + O_PGE, ws + O_PSUM);
  k_pf1<<<Bc, 64, 0, stream>>>(ws + O_PSUM, pf_W1, pf_b1,
      ws + O_YPF1, ws + O_PF1S, ws + O_PF1Q);
  k_fin<<<1, 64, 0, stream>>>(ws + O_PF1S, ws + O_PF1Q, 1.f / Bc,
      ws + O_PF1M, ws + O_PF1R, 64);
  k_bnrelu_gemv<<<Bc, 64, 0, stream>>>(ws + O_YPF1, ws + O_PF1M, ws + O_PF1R,
      pf_g1, pf_be1, pf_W2, pf_b2, ws + O_YPF2, ws + O_PF2S, ws + O_PF2Q);
  k_fin<<<1, 64, 0, stream>>>(ws + O_PF2S, ws + O_PF2Q, 1.f / Bc,
      ws + O_PF2M, ws + O_PF2R, 64);
  k_bn_apply_stats<<<Bc, 64, 0, stream>>>(ws + O_YPF2, ws + O_PF2M, ws + O_PF2R,
      pf_g2, pf_be2, ws + O_FUSED, ws + O_FUSS, ws + O_FUSQ);

  // 4) bnp params from decomposed stats, pre-scaled activations, bf16 weights
  k_zparams<<<1, 64, 0, stream>>>(ws + O_BASES, ws + O_BASEQ,
      ws + O_FUSS, ws + O_FUSQ, bnp_g, bnp_b, ws + O_ZSC, ws + O_ZSH);
  k_prepA<<<Gc + Bc, 64, 0, stream>>>(ws + O_BASE, ws + O_FUSED,
      ws + O_ZSC, ws + O_ZSH, ws + O_BASEA, ws + O_FUSEDA);
  k_wconv<<<64, 256, 0, stream>>>(r_W1, r_W2, W1bf, W2bf);

  // 5) heavy fused MLP, three WMMA passes (recompute > store; L2-resident inputs)
  dim3 grid(NTG, Bc);
  k_mlp<0><<<grid, 256, 0, stream>>>(ws + O_BASEA, ws + O_FUSEDA, W1bf, W2bf,
      r_b1, ws + O_TSC, ws + O_TSH, r_b2, ws + O_OSC, ws + O_OSH,
      indv_w1, indv_b1, ws + O_S1S, ws + O_S1Q, ws + O_S2S, ws + O_S2Q, ws + O_WVEC);
  k_fin_slots<<<1, 128, 0, stream>>>(ws + O_S1S, ws + O_S1Q, 128,
      1.f / (float)Nrows, r_g1, r_be1, ws + O_TSC, ws + O_TSH);
  k_mlp<1><<<grid, 256, 0, stream>>>(ws + O_BASEA, ws + O_FUSEDA, W1bf, W2bf,
      r_b1, ws + O_TSC, ws + O_TSH, r_b2, ws + O_OSC, ws + O_OSH,
      indv_w1, indv_b1, ws + O_S1S, ws + O_S1Q, ws + O_S2S, ws + O_S2Q, ws + O_WVEC);
  k_fin_slots<<<1, 64, 0, stream>>>(ws + O_S2S, ws + O_S2Q, 64,
      1.f / (float)Nrows, r_g2, r_be2, ws + O_OSC, ws + O_OSH);
  k_mlp<2><<<grid, 256, 0, stream>>>(ws + O_BASEA, ws + O_FUSEDA, W1bf, W2bf,
      r_b1, ws + O_TSC, ws + O_TSH, r_b2, ws + O_OSC, ws + O_OSH,
      indv_w1, indv_b1, ws + O_S1S, ws + O_S1Q, ws + O_S2S, ws + O_S2Q, ws + O_WVEC);

  // 6) cross_gene_state MLP
  k_cross1<<<Bc, 256, 0, stream>>>(ws + O_WVEC, c_W1, c_b1,
      ws + O_YC1, ws + O_C1S, ws + O_C1Q);
  k_fin<<<1, 64, 0, stream>>>(ws + O_C1S, ws + O_C1Q, 1.f / Bc,
      ws + O_C1M, ws + O_C1R, 64);
  k_bnrelu_gemv<<<Bc, 64, 0, stream>>>(ws + O_YC1, ws + O_C1M, ws + O_C1R,
      c_g1, c_be1, c_W2, c_b2, ws + O_YC2, ws + O_C2S, ws + O_C2Q);
  k_fin_affine<<<1, 64, 0, stream>>>(ws + O_C2S, ws + O_C2Q, 1.f / Bc,
      c_g2, c_be2, ws + O_CSC, ws + O_CSH, 64);

  // 7) final per-gene combine + residual
  k_final<<<Bc, 256, 0, stream>>>(ws + O_WVEC, ws + O_YC2, ws + O_CSC, ws + O_CSH,
      indv_w2, indv_b2, expr, out);
}
